// MultiLevelROIVisualPrompt_14688788152366
// MI455X (gfx1250) — matrix operand order repaired and checked by reading, not compiled, not run on hardware
//
#include <hip/hip_runtime.h>

#define HW   9216      // 96*96
#define CTOT 2880      // 192+384+768+1536
#define KBOX 512
#define OUTK 7
#define TK   96        // K-tile in bf16 elements (3 WMMA k-steps)

typedef __attribute__((ext_vector_type(16))) __bf16 v16bf;
typedef __attribute__((ext_vector_type(8)))  float  v8f;
typedef __attribute__((ext_vector_type(4)))  unsigned int u32x4;
typedef __attribute__((ext_vector_type(8)))  int          i32x8;
typedef __attribute__((ext_vector_type(4)))  int          i32x4;

static __device__ __forceinline__ unsigned short f2bf(float x) {
    unsigned u = __builtin_bit_cast(unsigned, x);
    unsigned r = u + 0x7fffu + ((u >> 16) & 1u);   // round-to-nearest-even
    return (unsigned short)(r >> 16);
}

// -------- torchvision ROI-Align axis sampling (aligned=False, adaptive grid) --------
static __device__ __forceinline__ void axis_taps(float lo, float b, int grid, int p, int s,
                                                 int& i0, int& i1, float& w0, float& w1) {
    float coord = lo + (float)p * b + ((float)s + 0.5f) * (b / (float)grid);
    bool  oob   = (coord < -1.0f) || (coord > 96.0f);
    float c     = fmaxf(coord, 0.0f);
    int   f0    = (int)floorf(c);
    bool  hi    = f0 >= 95;
    i0 = hi ? 95 : f0;
    i1 = hi ? 95 : f0 + 1;
    float f = hi ? 0.0f : (c - (float)i0);
    float m = oob ? 0.0f : 1.0f;
    w0 = (1.0f - f) * m;
    w1 = f * m;
}

// Kernel 1: densify ROI-mean weight matrix into bf16 [KBOX, HW], one block per box.
__global__ __launch_bounds__(256) void build_wmat(const float* __restrict__ boxes,
                                                  unsigned short* __restrict__ wmatBf) {
    __shared__ float w[HW];                       // 36 KB of the 320 KB WGP LDS
    const int k = blockIdx.x;
    for (int i = threadIdx.x; i < HW; i += 256) w[i] = 0.0f;
    __syncthreads();

    const float x1 = boxes[k * 4 + 0], y1 = boxes[k * 4 + 1];
    const float x2 = boxes[k * 4 + 2], y2 = boxes[k * 4 + 3];
    const float rw = fmaxf(x2 - x1, 1.0f), rh = fmaxf(y2 - y1, 1.0f);
    const int   gw = (int)fminf(fmaxf(ceilf(rw / 7.0f), 1.0f), 8.0f);
    const int   gh = (int)fminf(fmaxf(ceilf(rh / 7.0f), 1.0f), 8.0f);
    const float bw = rw / 7.0f, bh = rh / 7.0f;
    const float norm = 1.0f / ((float)(gh * gw) * 49.0f);
    const int nS = 49 * gh * gw;

    for (int idx = threadIdx.x; idx < nS; idx += 256) {
        int t = idx;
        const int sx = t % gw; t /= gw;
        const int sy = t % gh; t /= gh;
        const int px = t % OUTK;
        const int py = t / OUTK;
        int yi0, yi1, xi0, xi1; float wy0, wy1, wx0, wx1;
        axis_taps(y1, bh, gh, py, sy, yi0, yi1, wy0, wy1);
        axis_taps(x1, bw, gw, px, sx, xi0, xi1, wx0, wx1);
        atomicAdd(&w[yi0 * 96 + xi0], wy0 * wx0 * norm);   // ds_add_f32
        atomicAdd(&w[yi0 * 96 + xi1], wy0 * wx1 * norm);
        atomicAdd(&w[yi1 * 96 + xi0], wy1 * wx0 * norm);
        atomicAdd(&w[yi1 * 96 + xi1], wy1 * wx1 * norm);
    }
    __syncthreads();
    for (int i = threadIdx.x; i < HW; i += 256)
        wmatBf[(size_t)k * HW + i] = f2bf(w[i]);
}

// Kernel 2: bilinear-upsample (half-pixel) levels 1..3 to 96x96, concat, fp32 -> bf16.
__global__ __launch_bounds__(256) void pack_feat(const float* __restrict__ f0,
                                                 const float* __restrict__ f1,
                                                 const float* __restrict__ f2,
                                                 const float* __restrict__ f3,
                                                 unsigned short* __restrict__ featBf) {
    const long long gid = (long long)blockIdx.x * 256 + threadIdx.x;
    if (gid >= (long long)CTOT * HW) return;
    const int c   = (int)(gid / HW);
    const int pix = (int)(gid % HW);
    const int y = pix / 96, x = pix % 96;
    float val;
    if (c < 192) {
        val = f0[(size_t)c * HW + pix];
    } else {
        const float* src; int S, cs;
        if (c < 576)       { src = f1; S = 48; cs = c - 192;  }
        else if (c < 1344) { src = f2; S = 24; cs = c - 576;  }
        else               { src = f3; S = 12; cs = c - 1344; }
        const float scale = (float)S * (1.0f / 96.0f);
        const float sx = ((float)x + 0.5f) * scale - 0.5f;
        const float sy = ((float)y + 0.5f) * scale - 0.5f;
        const int x0 = (int)floorf(sx), y0 = (int)floorf(sy);
        const float fx = sx - (float)x0, fy = sy - (float)y0;
        const int x0c = min(max(x0, 0), S - 1), x1c = min(max(x0 + 1, 0), S - 1);
        const int y0c = min(max(y0, 0), S - 1), y1c = min(max(y0 + 1, 0), S - 1);
        const float* p = src + (size_t)cs * S * S;
        const float v00 = p[y0c * S + x0c], v01 = p[y0c * S + x1c];
        const float v10 = p[y1c * S + x0c], v11 = p[y1c * S + x1c];
        val = (1.0f - fy) * ((1.0f - fx) * v00 + fx * v01)
            +          fy * ((1.0f - fx) * v10 + fx * v11);
    }
    featBf[gid] = f2bf(val);
}

// TDM: one tensor_load_to_lds moves a [rows x TK] bf16 tile (row stride HW elements)
// from global memory into LDS at lds_addr. 2-D descriptor, groups 2/3/4 zero.
// clang-23 arity: (u32x4 g0, i32x8 g1, i32x4 g2, i32x4 g3, i32x8 g4, i32 cpol).
static __device__ __forceinline__ void tdm_load_tile(unsigned lds_addr, const void* gptr,
                                                     int rows) {
    const unsigned long long ga = (unsigned long long)(size_t)gptr;
    u32x4 g0;
    g0[0] = 1u;                                   // count=1, is_restore=0, gather off
    g0[1] = lds_addr;                             // D#.lds_addr (bytes)
    g0[2] = (unsigned)ga;                         // D#.global_addr[31:0]
    g0[3] = (unsigned)(ga >> 32) | (2u << 30);    // global_addr[56:32] | type=2
    i32x8 g1;
    g1[0] = (int)(1u << 16);                      // workgroup_mask=0, data_size=1 (2B)
    g1[1] = (int)((unsigned)TK << 16);            // tensor_dim0[15:0] = TK
    g1[2] = (int)((unsigned)rows << 16);          // tensor_dim0[31:16]=0, tensor_dim1[15:0]=rows
    g1[3] = (int)((unsigned)TK << 16);            // tensor_dim1[31:16]=0, tile_dim0 = TK
    g1[4] = rows;                                 // tile_dim1 = rows, tile_dim2 = 0
    g1[5] = HW;                                   // tensor_dim0_stride[31:0] = HW
    g1[6] = 0;                                    // stride high bits / dim1_stride lo
    g1[7] = 0;                                    // dim1_stride hi (unused, 2-D)
    const i32x4 z4 = { 0, 0, 0, 0 };
    const i32x8 z8 = { 0, 0, 0, 0, 0, 0, 0, 0 };
    __builtin_amdgcn_tensor_load_to_lds(g0, g1, z4, z4, z8, 0);
}

// Kernel 3: out[512,2880] = Wmat[512,9216](bf16) x feat[2880,9216](bf16)^T, fp32 accum.
// Tile 64(M) x 96(N), TK=96 (3 WMMA k-steps); 6 waves (2x3), each wave -> 32x32.
// Double-buffered LDS; tiles fetched by the Tensor Data Mover (wave 0 issues the two
// descriptors per K-step, TENSORcnt-tracked), WMMAs consume the current buffer.
// K loop kept rolled (unroll 1) with runtime buffer offsets to minimize RA churn.
__global__ __launch_bounds__(192) void roi_gemm(const unsigned short* __restrict__ A,
                                                const unsigned short* __restrict__ B,
                                                float* __restrict__ out) {
    __shared__ unsigned short sA[2 * 64 * TK];   // 2 x 12 KB
    __shared__ unsigned short sB[2 * 96 * TK];   // 2 x 18 KB
    const int tid   = threadIdx.x;
    const int lane  = tid & 31;
    const int wave  = tid >> 5;
    const int waveM = wave / 3;              // 0..1
    const int waveN = wave % 3;              // 0..2
    const int half  = lane >> 4;             // lane group (0: lanes 0-15, 1: 16-31)
    const int lm    = lane & 15;
    const int mBase = blockIdx.y * 64;
    const int nBase = blockIdx.x * 96;

    const unsigned short* gA = &A[(size_t)mBase * HW];   // tile row origin, += kt per step
    const unsigned short* gB = &B[(size_t)nBase * HW];
    const unsigned ldsA = (unsigned)(size_t)(void*)&sA[0];
    const unsigned ldsB = (unsigned)(size_t)(void*)&sB[0];

    // Per-wave fragment base indices (halves), buffer parity added per iteration.
    const int aRow0 = (waveM * 32 + lm) * TK;            // mt=0 rows; mt=1 adds 16*TK
    const int bCol0 = (waveN * 32 + lm) * TK;            // nt=0 cols; nt=1 adds 16*TK

    v8f zero = {};
    v8f acc[2][2] = { { zero, zero }, { zero, zero } };

    // Prologue: TDM-stage tile 0 into buffer 0.
    if (wave == 0) {
        tdm_load_tile(ldsA, gA, 64);
        tdm_load_tile(ldsB, gB, 96);
        __builtin_amdgcn_s_wait_tensorcnt(0);
    }
    __syncthreads();

    const int T = HW / TK;                       // 96 K-steps
    #pragma unroll 1
    for (int t = 0; t < T; ++t) {
        const int offA = (t & 1) * (64 * TK);    // current buffer offsets (halves)
        const int offB = (t & 1) * (96 * TK);
        const bool has_next = (t + 1 < T);
        if (has_next && wave == 0) {             // async DMA of next tiles into buf^1
            const int kn = (t + 1) * TK;
            const unsigned nA = ldsA + (unsigned)(((t + 1) & 1) * (64 * TK * 2));
            const unsigned nB = ldsB + (unsigned)(((t + 1) & 1) * (96 * TK * 2));
            tdm_load_tile(nA, gA + kn, 64);
            tdm_load_tile(nB, gB + kn, 96);
        }

        // Compute: 3 k-substeps x 4 WMMAs from the current LDS buffer.
        union Frag { uint4 q[2]; v16bf v; };
        #pragma unroll
        for (int ks = 0; ks < 3; ++ks) {
            Frag a[2], b[2];
            #pragma unroll
            for (int mt = 0; mt < 2; ++mt) {
                const unsigned short* p = &sA[offA + aRow0 + mt * (16 * TK) + ks * 32];
                a[mt].q[0] = *(const uint4*)&p[half * 8];        // K = half*8 + 0..7
                a[mt].q[1] = *(const uint4*)&p[16 + half * 8];   // K = 16 + half*8 + 0..7
            }
            #pragma unroll
            for (int nt = 0; nt < 2; ++nt) {
                const unsigned short* p = &sB[offB + bCol0 + nt * (16 * TK) + ks * 32];
                b[nt].q[0] = *(const uint4*)&p[half * 8];        // K = half*8 + 0..7
                b[nt].q[1] = *(const uint4*)&p[16 + half * 8];   // K = 16 + half*8 + 0..7
            }
            #pragma unroll
            for (int mt = 0; mt < 2; ++mt)
                #pragma unroll
                for (int nt = 0; nt < 2; ++nt)
                    acc[mt][nt] = __builtin_amdgcn_wmma_f32_16x16x32_bf16(
                        false, a[mt].v, false, b[nt].v, (short)0, acc[mt][nt], false, false);
        }

        if (has_next) {
            if (wave == 0) __builtin_amdgcn_s_wait_tensorcnt(0);  // DMA complete
            __syncthreads();                                      // publish buf^1
        }
    }

    // Epilogue: D layout — lane lm = column, VGPR r -> row (half*8 + r).
    #pragma unroll
    for (int mt = 0; mt < 2; ++mt) {
        #pragma unroll
        for (int nt = 0; nt < 2; ++nt) {
            const int col   = nBase + waveN * 32 + nt * 16 + lm;
            const int rbase = mBase + waveM * 32 + mt * 16 + half * 8;
            #pragma unroll
            for (int r = 0; r < 8; ++r)
                out[(size_t)(rbase + r) * CTOT + col] = acc[mt][nt][r];
        }
    }
}

extern "C" void kernel_launch(void* const* d_in, const int* in_sizes, int n_in,
                              void* d_out, int out_size, void* d_ws, size_t ws_size,
                              hipStream_t stream) {
    const float* feat0 = (const float*)d_in[0];
    const float* feat1 = (const float*)d_in[1];
    const float* feat2 = (const float*)d_in[2];
    const float* feat3 = (const float*)d_in[3];
    const float* boxes = (const float*)d_in[4];

    unsigned short* featBf = (unsigned short*)d_ws;                             // 53.1 MB
    unsigned short* wmatBf = (unsigned short*)((char*)d_ws
                             + (size_t)CTOT * HW * sizeof(unsigned short));     // +9.44 MB
    float* out = (float*)d_out;                                                 // [512, 2880]

    build_wmat<<<KBOX, 256, 0, stream>>>(boxes, wmatBf);

    const long long total = (long long)CTOT * HW;
    const int blocks = (int)((total + 255) / 256);
    pack_feat<<<blocks, 256, 0, stream>>>(feat0, feat1, feat2, feat3, featBf);

    dim3 grid(CTOT / 96, KBOX / 64);   // 30 x 8
    roi_gemm<<<grid, 192, 0, stream>>>(wmatBf, featBf, out);
}
